// SelfAttention_54202487276081
// MI455X (gfx1250) — matrix operand order repaired and checked by reading, not compiled, not run on hardware
//
#include <hip/hip_runtime.h>
#include <hip/hip_bf16.h>
#include <math.h>

typedef __attribute__((ext_vector_type(16))) _Float16 v16h;
typedef __attribute__((ext_vector_type(8)))  _Float16 v8h;
typedef __attribute__((ext_vector_type(8)))  float    v8f;
typedef __attribute__((ext_vector_type(4)))  int      v4i;

#define BM 128
#define BN 128
#define BK 32
#define LDT 40   // LDS row stride in halves: 80B, 16B-aligned, spreads banks

// CDNA5 async global->LDS path (ASYNCcnt-tracked), if this toolchain declares it.
#if defined(__has_builtin)
#if __has_builtin(__builtin_amdgcn_global_load_async_to_lds_b128) && \
    __has_builtin(__builtin_amdgcn_s_wait_asynccnt)
#define HAVE_ASYNC_LDS 1
#endif
#endif
#ifndef HAVE_ASYNC_LDS
#define HAVE_ASYNC_LDS 0
#endif

#if HAVE_ASYNC_LDS
// builtin prototype (from clang diagnostics): (v4i as1*, v4i as3*, imm, imm)
#define GLOBAL_V4I(p) ((__attribute__((address_space(1))) v4i*)(p))
#define LDS_V4I(p)    ((__attribute__((address_space(3))) v4i*)(p))
#endif

enum { EPI_PROJ = 0, EPI_SCORES = 1, EPI_OUT = 2 };

union FragU { v16h v; v8h h[2]; };

static __device__ __forceinline__ int imin(int a, int b) { return a < b ? a : b; }

// D = A * B(+mask/scale/bias), A:[M,K] row-major, B either [N,K] (B_NT=true,
// K-contiguous per output col: nn.Linear weights / K-matrix for Q.K^T) or
// [K,N] (B_NT=false, transposed into LDS at stage time: V).
// Double-buffered LDS; pure-copy f16 stages use GLOBAL_LOAD_ASYNC_TO_LDS_B128
// so tile k+1 streams into LDS while tile k runs through the WMMA pipe.
template<bool A_F32, bool B_F32, bool B_NT, int EPI, bool KLIMIT>
__global__ __launch_bounds__(256)
void gemm_wmma_f16(const void* __restrict__ Ag, const void* __restrict__ Bg,
                   const float* __restrict__ bias, void* __restrict__ Og,
                   int M, int N, int K, int lda, int ldb, int ldo, float scale)
{
    __shared__ alignas(16) _Float16 As[2][BM * LDT];
    __shared__ alignas(16) _Float16 Bs[2][BN * LDT];

    const int t  = threadIdx.x;
    const int n0 = blockIdx.x * BN;
    const int m0 = blockIdx.y * BM;

    // Fully-masked causal tile: no compute, just fill.
    if (EPI == EPI_SCORES && n0 > m0 + (BM - 1)) {
        float* outf = (float*)Og;
        for (int e = t; e < BM * BN; e += 256) {
            int r = e >> 7, c = e & 127;
            outf[(size_t)(m0 + r) * ldo + (n0 + c)] = -1e30f;
        }
        return;
    }

    const int wave   = t >> 5;          // 8 waves (wave32)
    const int lane   = t & 31;
    const int waveM  = wave & 3;        // 4 waves along M  -> 32 rows each
    const int waveN  = wave >> 2;       // 2 waves along N  -> 64 cols each
    const int laneLo = lane & 15;
    const int laneHi = lane >> 4;
    const int rA = waveM * 32;
    const int cB = waveN * 64;

    v8f zero = {0.f,0.f,0.f,0.f,0.f,0.f,0.f,0.f};
    v8f acc[2][4];
#pragma unroll
    for (int mi = 0; mi < 2; ++mi)
#pragma unroll
        for (int ni = 0; ni < 4; ++ni) acc[mi][ni] = zero;

    int kEnd = K;
    if (KLIMIT) kEnd = imin(K, m0 + BM);   // causal: keys beyond query tile are zero-prob

    // staging index maps
    const int sRow   = t >> 1;            // 0..127
    const int sCol   = (t & 1) * 16;      // 0 / 16
    const int vKRow  = t >> 3;            // 0..31   (B transpose staging)
    const int vNBase = (t & 7) * 16;      // 0..112

    // per-thread global base pointers (k0 added per stage)
    const float*    Af = (const float*)Ag    + (size_t)(m0 + sRow) * lda + sCol;
    const _Float16* Ah = (const _Float16*)Ag + (size_t)(m0 + sRow) * lda + sCol;
    const float*    Bf = (const float*)Bg    + (size_t)(n0 + sRow) * ldb + sCol;
    const _Float16* Bh = (const _Float16*)Bg + (size_t)(n0 + sRow) * ldb + sCol;
    const _Float16* Bt = (const _Float16*)Bg + (size_t)vKRow * ldb + (n0 + vNBase);

    // async ops issued per thread per stage (compile-time, for s_wait_asynccnt imm)
    constexpr int AOPS = ((!A_F32) ? 2 : 0) + ((B_NT && !B_F32) ? 2 : 0);
    (void)AOPS;

    auto stageA = [&](int buf, int k0) {
        _Float16* dst = &As[buf][sRow * LDT + sCol];
        if (A_F32) {
            const float4* ap4 = (const float4*)(Af + k0);
            _Float16 tmp[16];
#pragma unroll
            for (int j = 0; j < 4; ++j) {
                float4 f = ap4[j];
                tmp[4*j+0] = (_Float16)f.x; tmp[4*j+1] = (_Float16)f.y;
                tmp[4*j+2] = (_Float16)f.z; tmp[4*j+3] = (_Float16)f.w;
            }
            *(v8h*)dst       = *(v8h*)&tmp[0];
            *(v8h*)(dst + 8) = *(v8h*)&tmp[8];
        } else {
            const _Float16* ap = Ah + k0;
#if HAVE_ASYNC_LDS
            __builtin_amdgcn_global_load_async_to_lds_b128(
                GLOBAL_V4I(ap), LDS_V4I(dst), 0, 0);
            __builtin_amdgcn_global_load_async_to_lds_b128(
                GLOBAL_V4I(ap + 8), LDS_V4I(dst + 8), 0, 0);
#else
            _Float16 tmp[16];
            *(v8h*)&tmp[0] = *(const v8h*)ap;
            *(v8h*)&tmp[8] = *(const v8h*)(ap + 8);
            *(v8h*)dst       = *(v8h*)&tmp[0];
            *(v8h*)(dst + 8) = *(v8h*)&tmp[8];
#endif
        }
    };

    auto stageB = [&](int buf, int k0) {
        if (B_NT) {
            _Float16* dst = &Bs[buf][sRow * LDT + sCol];
            if (B_F32) {
                const float4* bp4 = (const float4*)(Bf + k0);
                _Float16 tmp[16];
#pragma unroll
                for (int j = 0; j < 4; ++j) {
                    float4 f = bp4[j];
                    tmp[4*j+0] = (_Float16)f.x; tmp[4*j+1] = (_Float16)f.y;
                    tmp[4*j+2] = (_Float16)f.z; tmp[4*j+3] = (_Float16)f.w;
                }
                *(v8h*)dst       = *(v8h*)&tmp[0];
                *(v8h*)(dst + 8) = *(v8h*)&tmp[8];
            } else {
                const _Float16* bp = Bh + k0;
#if HAVE_ASYNC_LDS
                __builtin_amdgcn_global_load_async_to_lds_b128(
                    GLOBAL_V4I(bp), LDS_V4I(dst), 0, 0);
                __builtin_amdgcn_global_load_async_to_lds_b128(
                    GLOBAL_V4I(bp + 8), LDS_V4I(dst + 8), 0, 0);
#else
                _Float16 tmp[16];
                *(v8h*)&tmp[0] = *(const v8h*)bp;
                *(v8h*)&tmp[8] = *(const v8h*)(bp + 8);
                *(v8h*)dst       = *(v8h*)&tmp[0];
                *(v8h*)(dst + 8) = *(v8h*)&tmp[8];
#endif
            }
        } else {
            // B is [K][N] (V): coalesced load, transposed scatter to LDS [N][K]
            const _Float16* bp = Bt + (size_t)k0 * ldb;
            _Float16 tmp[16];
            *(v8h*)&tmp[0] = *(const v8h*)bp;
            *(v8h*)&tmp[8] = *(const v8h*)(bp + 8);
#pragma unroll
            for (int j = 0; j < 16; ++j)
                Bs[buf][(vNBase + j) * LDT + vKRow] = tmp[j];
        }
    };

    const int kTiles = kEnd / BK;

    // prologue: stage tile 0 into buffer 0
    stageA(0, 0);
    stageB(0, 0);

    for (int it = 0; it < kTiles; ++it) {
        const int  cur     = it & 1;
        const bool hasNext = (it + 1) < kTiles;

        // stream tile it+1 into the other buffer while we compute tile it
        if (hasNext) {
            stageA(cur ^ 1, (it + 1) * BK);
            stageB(cur ^ 1, (it + 1) * BK);
        }

#if HAVE_ASYNC_LDS
        if (AOPS > 0) {
            // in-order completion: waiting down to the newer stage's op count
            // guarantees the current stage's copies have landed in LDS
            if (hasNext) __builtin_amdgcn_s_wait_asynccnt(AOPS);
            else         __builtin_amdgcn_s_wait_asynccnt(0);
        }
#endif
        __syncthreads();   // everyone's tile-it stage visible (ds + async)

        // ---- fragments per ISA layout, then 2x4 WMMA ----
        FragU a[2], b[4];
#pragma unroll
        for (int mi = 0; mi < 2; ++mi) {
            int m = rA + mi * 16 + laneLo;
            // A 16-bit 16x32: lane half h holds K = 8h+0..7 and 16+8h+0..7
            a[mi].h[0] = *(const v8h*)&As[cur][m * LDT + 8 * laneHi];
            a[mi].h[1] = *(const v8h*)&As[cur][m * LDT + 16 + 8 * laneHi];
        }
#pragma unroll
        for (int ni = 0; ni < 4; ++ni) {
            int n = cB + ni * 16 + laneLo;
            // B 16-bit 32x16: lane half h holds K = 8h+0..7 and 16+8h+0..7 for column N
            b[ni].h[0] = *(const v8h*)&Bs[cur][n * LDT + 8 * laneHi];
            b[ni].h[1] = *(const v8h*)&Bs[cur][n * LDT + 16 + 8 * laneHi];
        }
#pragma unroll
        for (int mi = 0; mi < 2; ++mi)
#pragma unroll
            for (int ni = 0; ni < 4; ++ni)
                acc[mi][ni] = __builtin_amdgcn_wmma_f32_16x16x32_f16(
                    false, a[mi].v, false, b[ni].v, (short)0, acc[mi][ni],
                    false, false);

        __syncthreads();   // tile-it reads done before it+2 overwrites buffer
    }

    // ---- epilogue: C/D layout M = r + 8*(lane/16), N = lane%16 ----
#pragma unroll
    for (int mi = 0; mi < 2; ++mi) {
#pragma unroll
        for (int ni = 0; ni < 4; ++ni) {
#pragma unroll
            for (int r = 0; r < 8; ++r) {
                int gm = m0 + rA + mi * 16 + r + 8 * laneHi;
                int gn = n0 + cB + ni * 16 + laneLo;
                float v = acc[mi][ni][r];
                if (EPI == EPI_PROJ) {
                    ((_Float16*)Og)[(size_t)gm * ldo + gn] =
                        (_Float16)(v + bias[gn]);
                } else if (EPI == EPI_SCORES) {
                    float s = v * scale;
                    if (gn > gm) s = -1e30f;   // causal mask
                    ((float*)Og)[(size_t)gm * ldo + gn] = s;
                } else {
                    ((float*)Og)[(size_t)gm * ldo + gn] = v;
                }
            }
        }
    }
}

// One block per row: max, sum(exp), normalize -> f16 probabilities.
__global__ __launch_bounds__(256)
void softmax_rows(const float* __restrict__ S_, _Float16* __restrict__ P, int n)
{
    __shared__ float red[256];
    const size_t base = (size_t)blockIdx.x * n;
    const int t = threadIdx.x;

    float mx = -3.4e38f;
    for (int k = t; k < n; k += 256) mx = fmaxf(mx, S_[base + k]);
    red[t] = mx; __syncthreads();
    for (int s = 128; s > 0; s >>= 1) {
        if (t < s) red[t] = fmaxf(red[t], red[t + s]);
        __syncthreads();
    }
    const float rowmax = red[0];
    __syncthreads();

    float sum = 0.f;
    for (int k = t; k < n; k += 256) sum += __expf(S_[base + k] - rowmax);
    red[t] = sum; __syncthreads();
    for (int s = 128; s > 0; s >>= 1) {
        if (t < s) red[t] += red[t + s];
        __syncthreads();
    }
    const float inv = 1.0f / red[0];

    for (int k = t; k < n; k += 256)
        P[base + k] = (_Float16)(__expf(S_[base + k] - rowmax) * inv);
}

extern "C" void kernel_launch(void* const* d_in, const int* in_sizes, int n_in,
                              void* d_out, int out_size, void* d_ws, size_t ws_size,
                              hipStream_t stream)
{
    (void)in_sizes; (void)n_in; (void)out_size; (void)ws_size;
    constexpr int B = 4, S = 2048, E = 1024;

    const float* x  = (const float*)d_in[0];
    const float* Wq = (const float*)d_in[1];
    const float* bq = (const float*)d_in[2];
    const float* Wk = (const float*)d_in[3];
    const float* bk = (const float*)d_in[4];
    const float* Wv = (const float*)d_in[5];
    const float* bv = (const float*)d_in[6];
    // d_in[7] = mask: causal tril, applied analytically in-kernel.

    // Workspace layout: Qh,Kh,Vh (f16, 16MB each) | scores (f32, 64MB) | P (f16, 32MB)
    const size_t nQKV = (size_t)B * S * E;     // 8M elems
    const size_t nSS  = (size_t)B * S * S;     // 16M elems
    _Float16* Qh = (_Float16*)d_ws;
    _Float16* Kh = Qh + nQKV;
    _Float16* Vh = Kh + nQKV;
    float*    Sc = (float*)(Vh + nQKV);
    _Float16* P  = (_Float16*)(Sc + nSS);
    float*    O  = (float*)d_out;

    const dim3 blk(256);
    const float inv_sqrtE = 1.0f / 32.0f;      // 1/sqrt(1024)

    // 1) QKV projections: [B*S,E] = x[B*S,E] @ W[E,E]^T + b  (W is [out,in])
    {
        dim3 grid(E / BN, (B * S) / BM);
        gemm_wmma_f16<true, true, true, EPI_PROJ, false><<<grid, blk, 0, stream>>>(
            x, Wq, bq, Qh, B * S, E, E, E, E, E, 1.0f);
        gemm_wmma_f16<true, true, true, EPI_PROJ, false><<<grid, blk, 0, stream>>>(
            x, Wk, bk, Kh, B * S, E, E, E, E, E, 1.0f);
        gemm_wmma_f16<true, true, true, EPI_PROJ, false><<<grid, blk, 0, stream>>>(
            x, Wv, bv, Vh, B * S, E, E, E, E, E, 1.0f);
    }

    // 2) scores[b] = Q[b] @ K[b]^T / sqrt(E), causal-masked (f32)
    for (int b = 0; b < B; ++b) {
        dim3 grid(S / BN, S / BM);
        gemm_wmma_f16<false, false, true, EPI_SCORES, false><<<grid, blk, 0, stream>>>(
            Qh + (size_t)b * S * E, Kh + (size_t)b * S * E, nullptr,
            Sc + (size_t)b * S * S, S, S, E, E, E, S, inv_sqrtE);
    }

    // 3) row softmax -> P (f16)
    softmax_rows<<<dim3(B * S), blk, 0, stream>>>(Sc, P, S);

    // 4) out[b] = P[b] @ V[b], k-loop clipped at causal limit
    for (int b = 0; b < B; ++b) {
        dim3 grid(E / BN, S / BM);
        gemm_wmma_f16<false, false, false, EPI_OUT, true><<<grid, blk, 0, stream>>>(
            P + (size_t)b * S * S, Vh + (size_t)b * S * E, nullptr,
            O + (size_t)b * S * E, S, E, S, S, E, E, 1.0f);
    }
}
